// BiLSTM_CRF_68152541053203
// MI455X (gfx1250) — matrix operand order repaired and checked by reading, not compiled, not run on hardware
//
#include <hip/hip_runtime.h>
#include <hip/hip_bf16.h>

typedef __attribute__((ext_vector_type(16))) _Float16 v16h;
typedef __attribute__((ext_vector_type(8)))  float    v8f;

#define BB   64
#define TT   256
#define CIN  25
#define CH   10
#define EMBI 320
#define HH   256
#define VV   100000
#define EE   300
#define KK   20

__device__ __forceinline__ float sigf(float x) { return 1.0f / (1.0f + __expf(-x)); }

// ---- WMMA operand packing per CDNA5 ISA 7.12.2 layouts -----------------
// A tile: 16x32 f16 (M x K). Row-major source, row stride = ldk halves.
__device__ __forceinline__ v16h load_a_tile(const _Float16* base, int ldk, int lane) {
  const int m  = lane & 15;
  const int hs = lane >> 4;           // K half-select
  v16h a;
#pragma unroll
  for (int v = 0; v < 16; ++v) {
    const int vg = v >> 1, hf = v & 1;
    const int k = (vg < 4) ? (hs * 8 + vg * 2 + hf)
                           : (16 + hs * 8 + (vg - 4) * 2 + hf);
    a[v] = base[m * ldk + k];
  }
  return a;
}
// B tile: 32x16 f16 (K x N) where B[k][n] = W[n][k]; W row-major, stride ldw.
__device__ __forceinline__ v16h load_b_tile_wT(const _Float16* wbase, int ldw, int lane) {
  const int n  = lane & 15;
  const int hs = lane >> 4;
  v16h b;
#pragma unroll
  for (int v = 0; v < 16; ++v) {
    const int k = (v < 8) ? (hs * 8 + v) : (16 + hs * 8 + (v - 8));  // table order, matches A
    b[v] = wbase[n * ldw + k];
  }
  return b;
}

// ---- kernel 0: char BiLSTM (tiny, VALU) --------------------------------
__global__ void char_lstm_kernel(const float* __restrict__ xc,
                                 const float* __restrict__ Wih_f, const float* __restrict__ Whh_f,
                                 const float* __restrict__ b_f,
                                 const float* __restrict__ Wih_b, const float* __restrict__ Whh_b,
                                 const float* __restrict__ b_b,
                                 float* __restrict__ ce) {
  const int bb = blockIdx.x;
  const int d  = blockIdx.y;
  const float* Wih  = d ? Wih_b : Wih_f;
  const float* Whh  = d ? Whh_b : Whh_f;
  const float* bias = d ? b_b   : b_f;
  __shared__ float h[CH], c[CH], gbuf[4 * CH];
  const int j = threadIdx.x;
  if (j < CH) { h[j] = 0.f; c[j] = 0.f; }
  __syncthreads();
  for (int t = 0; t < TT; ++t) {
    const int tt = d ? (TT - 1 - t) : t;
    if (j < 4 * CH) {
      float g = bias[j];
      const float* xr = xc + (bb * TT + tt) * CIN;
      for (int i = 0; i < CIN; ++i) g += Wih[j * CIN + i] * xr[i];
      for (int k = 0; k < CH; ++k)  g += Whh[j * CH + k] * h[k];
      gbuf[j] = g;
    }
    __syncthreads();
    if (j < CH) {
      const float gi = gbuf[j], gf = gbuf[CH + j], gg = gbuf[2 * CH + j], go = gbuf[3 * CH + j];
      const float cc = sigf(gf) * c[j] + sigf(gi) * tanhf(gg);
      c[j] = cc;
      const float hh = sigf(go) * tanhf(cc);
      h[j] = hh;
      ce[(bb * TT + tt) * (2 * CH) + d * CH + j] = hh;
    }
    __syncthreads();
  }
}

// ---- kernel 1: build x (char_emb ++ emb[token]) as f16 -----------------
__global__ void build_x16_kernel(const float* __restrict__ ce,
                                 const int* __restrict__ token,
                                 const float* __restrict__ emb,
                                 _Float16* __restrict__ x16) {
  const int idx = blockIdx.x * blockDim.x + threadIdx.x;
  if (idx >= BB * TT * EMBI) return;
  const int p = idx / EMBI, col = idx % EMBI;
  float v;
  if (col < 2 * CH) v = ce[p * (2 * CH) + col];
  else              v = emb[(long long)token[p] * EE + (col - 2 * CH)];
  x16[idx] = (_Float16)v;
}

// ---- kernel 2: f32 -> f16 weight conversion ----------------------------
__global__ void cvt_f16_kernel(const float* __restrict__ in, _Float16* __restrict__ out, int n) {
  const int i = blockIdx.x * blockDim.x + threadIdx.x;
  if (i < n) out[i] = (_Float16)in[i];
}

// ---- kernel 3: xg = x @ Wih^T + b  (one direction), WMMA ---------------
// grid (1024, 16), block 128 (4 waves); wave -> one 16x16 tile, K=320.
__global__ void xg_wmma_kernel(const _Float16* __restrict__ x16,
                               const _Float16* __restrict__ w16,   // (1024,320) row-major
                               const float* __restrict__ bias,     // (1024)
                               float* __restrict__ xg) {           // (16384,1024)
  const int lane = threadIdx.x & 31;
  const int wave = threadIdx.x >> 5;
  const int M0 = blockIdx.x * 16;
  const int N0 = (blockIdx.y * 4 + wave) * 16;
  v8f acc = {};
  for (int kc = 0; kc < EMBI; kc += 32) {
    v16h a = load_a_tile(x16 + (long long)M0 * EMBI + kc, EMBI, lane);
    v16h b = load_b_tile_wT(w16 + (long long)N0 * EMBI + kc, EMBI, lane);
    acc = __builtin_amdgcn_wmma_f32_16x16x32_f16(false, a, false, b, (short)0, acc, false, false);
  }
  const int n  = N0 + (lane & 15);
  const float bn = bias[n];
#pragma unroll
  for (int r = 0; r < 8; ++r) {
    const int m = M0 + r + 8 * (lane >> 4);
    xg[(long long)m * (4 * HH) + n] = acc[r] + bn;
  }
}

// ---- kernel 4: word LSTM recurrence, WMMA, cell update in registers ----
// grid (4, 2): (batch group of 16, direction). block 256 (8 waves).
// wave g owns gate tiles nt = 16*q + 2*g + u (q=gate, u=0/1) -> the same
// lane holds matching (i,f,g,o) for its (batch, hidden) pairs.
// Loop order: K-chunk outer so ONE A tile feeds 8 consecutive WMMAs.
__global__ void word_lstm_kernel(const float* __restrict__ xg_f,
                                 const float* __restrict__ xg_b,
                                 const _Float16* __restrict__ whh_f,  // (1024,256) f16
                                 const _Float16* __restrict__ whh_b,
                                 float* __restrict__ seq) {           // (64,256,512)
  const int bg = blockIdx.x;            // batch group: batches [16*bg, 16*bg+16)
  const int d  = blockIdx.y;            // direction
  const float*    xg  = d ? xg_b  : xg_f;
  const _Float16* whh = d ? whh_b : whh_f;
  const int bbase = bg * 16;
  const int lane = threadIdx.x & 31;
  const int g    = threadIdx.x >> 5;    // wave id 0..7
  const int tid  = threadIdx.x;

  __shared__ _Float16 hbuf[16 * HH];    // 8 KB: h state for 16 batches
  for (int i = tid; i < 16 * HH; i += blockDim.x) hbuf[i] = (_Float16)0.f;

  float creg[16];
#pragma unroll
  for (int r = 0; r < 16; ++r) creg[r] = 0.f;
  __syncthreads();

  for (int t = 0; t < TT; ++t) {
    const int tt = d ? (TT - 1 - t) : t;

    // Prefetch next timestep's xg slice (64 KB/block) behind this step's math.
    // 256 threads x 2 x 128B lines cover 16 rows x 4 KB. -> global_prefetch_b8
    if (t + 1 < TT) {
      const int ttn  = d ? (TT - 2 - t) : (t + 1);
      const int prow = bbase + (tid >> 4);                 // batch row
      const char* pb = (const char*)(xg + ((long long)prow * TT + ttn) * (4 * HH))
                       + (tid & 15) * 256;
      __builtin_prefetch(pb, 0, 3);
      __builtin_prefetch(pb + 128, 0, 3);
    }

    // --- init accumulators with xg[:,tt,:] ---
    v8f accs[8];
#pragma unroll
    for (int j = 0; j < 8; ++j) {
      const int q = j >> 1, u = j & 1;
      const int N0 = (16 * q + 2 * g + u) * 16;
      const int n  = N0 + (lane & 15);
#pragma unroll
      for (int r = 0; r < 8; ++r) {
        const int bglob = bbase + r + 8 * (lane >> 4);
        accs[j][r] = xg[((long long)bglob * TT + tt) * (4 * HH) + n];
      }
    }
    // --- gates += h @ Whh^T : one A tile per K-chunk, shared by 8 WMMAs ---
    for (int kc = 0; kc < HH; kc += 32) {
      const v16h a = load_a_tile(hbuf + kc, HH, lane);
#pragma unroll
      for (int j = 0; j < 8; ++j) {
        const int q = j >> 1, u = j & 1;
        const int N0 = (16 * q + 2 * g + u) * 16;
        v16h b = load_b_tile_wT(whh + (long long)N0 * HH + kc, HH, lane);
        accs[j] = __builtin_amdgcn_wmma_f32_16x16x32_f16(false, a, false, b, (short)0,
                                                         accs[j], false, false);
      }
    }
    __syncthreads();   // all waves done reading hbuf
    // --- LSTM cell update, fully register-resident c state ---
#pragma unroll
    for (int u = 0; u < 2; ++u) {
#pragma unroll
      for (int r = 0; r < 8; ++r) {
        const float gi = accs[0 * 2 + u][r];
        const float gf = accs[1 * 2 + u][r];
        const float gg = accs[2 * 2 + u][r];
        const float go = accs[3 * 2 + u][r];
        float c = sigf(gf) * creg[u * 8 + r] + sigf(gi) * tanhf(gg);
        creg[u * 8 + r] = c;
        const float h = sigf(go) * tanhf(c);
        const int bloc = r + 8 * (lane >> 4);        // local batch 0..15
        const int hx   = 32 * g + u * 16 + (lane & 15);
        hbuf[bloc * HH + hx] = (_Float16)h;
        seq[(((long long)(bbase + bloc)) * TT + tt) * (2 * HH) + d * HH + hx] = h;
      }
    }
    __syncthreads();   // h for step t complete
  }
}

// ---- kernel 5: em = seq @ Wtag^T + btag (K=20, tiny) -------------------
__global__ void tag_proj_kernel(const float* __restrict__ seq,
                                const float* __restrict__ Wtag,
                                const float* __restrict__ btag,
                                float* __restrict__ em) {
  const int idx = blockIdx.x * blockDim.x + threadIdx.x;
  if (idx >= BB * TT * KK) return;
  const int p = idx / KK, k = idx % KK;
  const float* sp = seq + (long long)p * (2 * HH);
  const float* wp = Wtag + k * (2 * HH);
  float s = btag[k];
  for (int i = 0; i < 2 * HH; ++i) s += sp[i] * wp[i];
  em[idx] = s;
}

// ---- kernel 6: CRF NLL (forward algorithm), one wave per batch ---------
__global__ void crf_kernel(const float* __restrict__ em, const int* __restrict__ tags,
                           const int* __restrict__ mask, const float* __restrict__ startv,
                           const float* __restrict__ endv, const float* __restrict__ trans,
                           float* __restrict__ out) {
  const int b = blockIdx.x;
  const int j = threadIdx.x;
  __shared__ float alpha[KK], nxt[KK];
  const float* eb = em + (long long)b * TT * KK;
  const int* tg = tags + b * TT;
  const int* mk = mask + b * TT;
  if (j < KK) alpha[j] = startv[j] + eb[j];
  __syncthreads();
  for (int t = 1; t < TT; ++t) {
    float v = 0.f;
    if (j < KK) {
      float mx = -3.4e38f;
      for (int i = 0; i < KK; ++i) mx = fmaxf(mx, alpha[i] + trans[i * KK + j]);
      float s = 0.f;
      for (int i = 0; i < KK; ++i) s += __expf(alpha[i] + trans[i * KK + j] - mx);
      v = mx + __logf(s) + eb[t * KK + j];
    }
    __syncthreads();
    if (j < KK) nxt[j] = v;
    __syncthreads();
    if (j < KK && mk[t] != 0) alpha[j] = nxt[j];
    __syncthreads();
  }
  if (j == 0) {
    float mx = -3.4e38f;
    for (int i = 0; i < KK; ++i) mx = fmaxf(mx, alpha[i] + endv[i]);
    float s = 0.f;
    for (int i = 0; i < KK; ++i) s += __expf(alpha[i] + endv[i] - mx);
    const float den = mx + __logf(s);
    float num = startv[tg[0]] + eb[tg[0]];
    int last = 0;
    for (int t = 1; t < TT; ++t) {
      if (mk[t] != 0) {
        num += trans[tg[t - 1] * KK + tg[t]] + eb[t * KK + tg[t]];
        last = t;
      }
    }
    num += endv[tg[last]];
    atomicAdd(out, den - num);   // loss = sum_b (den - num)
  }
}

extern "C" void kernel_launch(void* const* d_in, const int* in_sizes, int n_in,
                              void* d_out, int out_size, void* d_ws, size_t ws_size,
                              hipStream_t stream) {
  (void)in_sizes; (void)n_in; (void)ws_size;
  const float* char_x  = (const float*)d_in[0];
  const int*   token   = (const int*)  d_in[1];
  const int*   tags    = (const int*)  d_in[2];
  const int*   mask    = (const int*)  d_in[3];
  const float* emb     = (const float*)d_in[4];
  const float* cWih_f  = (const float*)d_in[5];
  const float* cWhh_f  = (const float*)d_in[6];
  const float* cb_f    = (const float*)d_in[7];
  const float* cWih_b  = (const float*)d_in[8];
  const float* cWhh_b  = (const float*)d_in[9];
  const float* cb_b    = (const float*)d_in[10];
  const float* wWih_f  = (const float*)d_in[11];
  const float* wWhh_f  = (const float*)d_in[12];
  const float* wb_f    = (const float*)d_in[13];
  const float* wWih_b  = (const float*)d_in[14];
  const float* wWhh_b  = (const float*)d_in[15];
  const float* wb_b    = (const float*)d_in[16];
  const float* Wtag    = (const float*)d_in[17];
  const float* btag    = (const float*)d_in[18];
  const float* start_t = (const float*)d_in[19];
  const float* end_t   = (const float*)d_in[20];
  const float* trans   = (const float*)d_in[21];
  float* out = (float*)d_out;

  // ---- workspace carve-up (all sizes 256B-aligned) ----
  char* ws = (char*)d_ws;
  size_t off = 0;
  float*    ce      = (float*)(ws + off);    off += (size_t)BB * TT * (2 * CH) * 4;   // 1.25 MB
  _Float16* x16     = (_Float16*)(ws + off); off += (size_t)BB * TT * EMBI * 2;       // 10 MB
  _Float16* wihf16  = (_Float16*)(ws + off); off += (size_t)(4 * HH) * EMBI * 2;
  _Float16* wihb16  = (_Float16*)(ws + off); off += (size_t)(4 * HH) * EMBI * 2;
  _Float16* whhf16  = (_Float16*)(ws + off); off += (size_t)(4 * HH) * HH * 2;
  _Float16* whhb16  = (_Float16*)(ws + off); off += (size_t)(4 * HH) * HH * 2;
  float*    xgf     = (float*)(ws + off);    off += (size_t)BB * TT * (4 * HH) * 4;   // 64 MB
  float*    xgb     = (float*)(ws + off);    off += (size_t)BB * TT * (4 * HH) * 4;   // 64 MB
  float*    seq     = (float*)(ws + off);    off += (size_t)BB * TT * (2 * HH) * 4;   // 32 MB
  float*    embuf   = (float*)(ws + off);    off += (size_t)BB * TT * KK * 4;         // 1.25 MB

  hipMemsetAsync(d_out, 0, (size_t)out_size * sizeof(float), stream);

  // char BiLSTM
  char_lstm_kernel<<<dim3(BB, 2), 64, 0, stream>>>(char_x, cWih_f, cWhh_f, cb_f,
                                                   cWih_b, cWhh_b, cb_b, ce);
  // x = [char_emb, emb[token]] as f16
  {
    const int n = BB * TT * EMBI;
    build_x16_kernel<<<(n + 255) / 256, 256, 0, stream>>>(ce, token, emb, x16);
  }
  // weight conversions to f16
  cvt_f16_kernel<<<((4 * HH * EMBI) + 255) / 256, 256, 0, stream>>>(wWih_f, wihf16, 4 * HH * EMBI);
  cvt_f16_kernel<<<((4 * HH * EMBI) + 255) / 256, 256, 0, stream>>>(wWih_b, wihb16, 4 * HH * EMBI);
  cvt_f16_kernel<<<((4 * HH * HH)   + 255) / 256, 256, 0, stream>>>(wWhh_f, whhf16, 4 * HH * HH);
  cvt_f16_kernel<<<((4 * HH * HH)   + 255) / 256, 256, 0, stream>>>(wWhh_b, whhb16, 4 * HH * HH);
  // input projections (big WMMA GEMMs)
  xg_wmma_kernel<<<dim3(BB * TT / 16, 16), 128, 0, stream>>>(x16, wihf16, wb_f, xgf);
  xg_wmma_kernel<<<dim3(BB * TT / 16, 16), 128, 0, stream>>>(x16, wihb16, wb_b, xgb);
  // recurrent word BiLSTM (WMMA per step, register-resident cell state)
  word_lstm_kernel<<<dim3(4, 2), 256, 0, stream>>>(xgf, xgb, whhf16, whhb16, seq);
  // emissions
  {
    const int n = BB * TT * KK;
    tag_proj_kernel<<<(n + 255) / 256, 256, 0, stream>>>(seq, Wtag, btag, embuf);
  }
  // CRF NLL -> scalar
  crf_kernel<<<BB, 32, 0, stream>>>(embuf, tags, mask, start_t, end_t, trans, out);
}
